// HMHSAGraph_4234837754307
// MI455X (gfx1250) — matrix operand order repaired and checked by reading, not compiled, not run on hardware
//
#include <hip/hip_runtime.h>
#include <hip/hip_bf16.h>
#include <math.h>

typedef __attribute__((ext_vector_type(16))) __bf16 v16bf;
typedef __attribute__((ext_vector_type(8)))  float  v8f;

#define N_HEADS  8
#define HEAD_DIM 8
#define EMBED    64
#define IN_FEAT  64
#define SCALING  0.35355339059327373f  /* 8^-0.5 */

__global__ void fill_kernel(float* __restrict__ p, float v, int n) {
  int i = blockIdx.x * blockDim.x + threadIdx.x;
  if (i < n) p[i] = v;
}

__device__ __forceinline__ void cvt8(v16bf& dst, int base, float4 a, float4 b) {
  dst[base + 0] = (__bf16)a.x; dst[base + 1] = (__bf16)a.y;
  dst[base + 2] = (__bf16)a.z; dst[base + 3] = (__bf16)a.w;
  dst[base + 4] = (__bf16)b.x; dst[base + 5] = (__bf16)b.y;
  dst[base + 6] = (__bf16)b.z; dst[base + 7] = (__bf16)b.w;
}

// ---------------------------------------------------------------------------
// QKV projection: y = x @ W^T + b via v_wmma_f32_16x16x32_bf16.
// One wave per 16-row tile of x; computes all of Q, K, V for that tile.
// A layout (16-bit, 16x32): lane half splits K into 8-blocks,
//   VGPR v -> K = (v/4)*16 + half*8 + (v%4)*2 (+ chunk*32)
//   => per lane, A needs floats [half*8, half*8+8) and [16+half*8, ...+8)
// B layout (16-bit, 32x16): lane = output column, K as for A: (v/4)*16 + half*8 + (v%4)*2 (+ chunk*32)
//   => per lane, B needs floats [half*8, half*8+8) and [16+half*8, ...+8)
// C/D layout: VGPR v, lane L -> M = (L/16)*8 + v, N = L%16
// ---------------------------------------------------------------------------
__global__ __launch_bounds__(128)
void qkv_kernel(const float* __restrict__ x,
                const float* __restrict__ Wq, const float* __restrict__ bq,
                const float* __restrict__ Wk, const float* __restrict__ bk,
                const float* __restrict__ Wv, const float* __restrict__ bv,
                float* __restrict__ Q, float* __restrict__ Kp, float* __restrict__ V,
                int N, int ntiles) {
  int wave = threadIdx.x >> 5;
  int tile = blockIdx.x * 4 + wave;
  if (tile >= ntiles) return;           // wave-uniform: EXEC stays all-ones
  int lane = threadIdx.x & 31;
  int half = lane >> 4;
  int lrow = lane & 15;
  int m0   = tile * 16;

  int row = m0 + lrow;
  if (row >= N) row = N - 1;            // clamp loads; stores masked below
  const float4* xr = (const float4*)(x + (long long)row * IN_FEAT);

  // A operands: chunk0 = K[0,32), chunk1 = K[32,64)
  v16bf a0, a1;
  {
    int q0 = half * 2;                  // float4 index of half*8
    cvt8(a0, 0, xr[q0],     xr[q0 + 1]);      // K = half*8 + [0..8)
    cvt8(a0, 8, xr[q0 + 4], xr[q0 + 5]);      // K = 16 + half*8 + [0..8)
    cvt8(a1, 0, xr[q0 + 8], xr[q0 + 9]);      // +32
    cvt8(a1, 8, xr[q0 +12], xr[q0 +13]);
  }

  const float* Ws[3]   = {Wq, Wk, Wv};
  const float* bs[3]   = {bq, bk, bv};
  float*       outs[3] = {Q, Kp, V};
  bool full = (m0 + 16 <= N);

  #pragma unroll
  for (int w = 0; w < 3; ++w) {
    #pragma unroll
    for (int nt = 0; nt < 4; ++nt) {
      const float4* wr = (const float4*)(Ws[w] + (long long)(nt * 16 + lrow) * IN_FEAT);
      v16bf b0v, b1v;
      {
        int q0 = half * 2;              // float4 index of half*8
        cvt8(b0v, 0, wr[q0],     wr[q0 + 1]); // K = half*8 + [0..8)
        cvt8(b0v, 8, wr[q0 + 4], wr[q0 + 5]); // K = 16 + half*8 + [0..8)
        cvt8(b1v, 0, wr[q0 + 8], wr[q0 + 9]); // +32
        cvt8(b1v, 8, wr[q0 +12], wr[q0 +13]);
      }
      float bias = bs[w][nt * 16 + lrow];
      v8f acc;
      #pragma unroll
      for (int i = 0; i < 8; ++i) acc[i] = bias;
      acc = __builtin_amdgcn_wmma_f32_16x16x32_bf16(false, a0, false, b0v,
                                                    (short)0, acc, false, false);
      acc = __builtin_amdgcn_wmma_f32_16x16x32_bf16(false, a1, false, b1v,
                                                    (short)0, acc, false, false);
      float* op = outs[w] + (long long)(m0 + half * 8) * EMBED + nt * 16 + lrow;
      if (full) {
        #pragma unroll
        for (int v = 0; v < 8; ++v) op[(long long)v * EMBED] = acc[v];
      } else {
        #pragma unroll
        for (int v = 0; v < 8; ++v)
          if (m0 + half * 8 + v < N) op[(long long)v * EMBED] = acc[v];
      }
    }
  }
}

// ---------------------------------------------------------------------------
// Per-(edge,head) scores + segment max over dst (native f32 max atomic).
// ---------------------------------------------------------------------------
__global__ void scores_kernel(const int* __restrict__ edge,
                              const float* __restrict__ Q, const float* __restrict__ K,
                              float* __restrict__ scores, float* __restrict__ segmax,
                              int E) {
  int t = blockIdx.x * blockDim.x + threadIdx.x;
  if (t >= E * N_HEADS) return;
  int e = t >> 3, h = t & 7;
  int src = edge[e], dst = edge[E + e];
  const float4* q = (const float4*)(Q + (long long)src * EMBED + h * HEAD_DIM);
  const float4* k = (const float4*)(K + (long long)dst * EMBED + h * HEAD_DIM);
  float4 q0 = q[0], q1 = q[1], k0 = k[0], k1 = k[1];
  float s = q0.x*k0.x + q0.y*k0.y + q0.z*k0.z + q0.w*k0.w
          + q1.x*k1.x + q1.y*k1.y + q1.z*k1.z + q1.w*k1.w;
  s *= SCALING;
  scores[t] = s;
  atomicMax(segmax + (long long)dst * N_HEADS + h, s);
}

__global__ void exp_kernel(const int* __restrict__ edge, float* __restrict__ scores,
                           const float* __restrict__ segmax, float* __restrict__ segsum,
                           int E) {
  int t = blockIdx.x * blockDim.x + threadIdx.x;
  if (t >= E * N_HEADS) return;
  int e = t >> 3, h = t & 7;
  int dst = edge[E + e];
  float m  = segmax[(long long)dst * N_HEADS + h];
  float ev = __expf(scores[t] - m);
  scores[t] = ev;
  atomicAdd(segsum + (long long)dst * N_HEADS + h, ev);
}

// normalize, write both attention outputs, and reduce head-mean per edge
__global__ void norm_kernel(const int* __restrict__ edge, float* __restrict__ attn1,
                            float* __restrict__ attn2, const float* __restrict__ segsum,
                            float* __restrict__ attn_mean, int E) {
  int t = blockIdx.x * blockDim.x + threadIdx.x;
  if (t >= E * N_HEADS) return;
  int e = t >> 3, h = t & 7;
  int dst = edge[E + e];
  float a = attn1[t] / (segsum[(long long)dst * N_HEADS + h] + 1e-16f);
  attn1[t] = a;
  attn2[t] = a;
  float s = a;                       // 8-lane tree within the wave32
  s += __shfl_xor(s, 1, 32);
  s += __shfl_xor(s, 2, 32);
  s += __shfl_xor(s, 4, 32);
  if (h == 0) attn_mean[e] = s * 0.125f;
}

// out[src] += attn_mean * Vflat[dst]; Vflat col c = V[:, (c%8)*8 + c/8]
__global__ void spmm_kernel(const int* __restrict__ edge, const float* __restrict__ V,
                            const float* __restrict__ attn_mean, float* __restrict__ out,
                            int E) {
  long long t = (long long)blockIdx.x * blockDim.x + threadIdx.x;
  if (t >= (long long)E * EMBED) return;
  int e = (int)(t >> 6), c = (int)(t & 63);
  int src = edge[e], dst = edge[E + e];
  float v = V[(long long)dst * EMBED + ((c & 7) * 8 + (c >> 3))];
  atomicAdd(out + (long long)src * EMBED + c, attn_mean[e] * v);
}

extern "C" void kernel_launch(void* const* d_in, const int* in_sizes, int n_in,
                              void* d_out, int out_size, void* d_ws, size_t ws_size,
                              hipStream_t stream) {
  const float* x    = (const float*)d_in[0];
  const int*   edge = (const int*)  d_in[1];
  const float* Wq   = (const float*)d_in[2];
  const float* bq   = (const float*)d_in[3];
  const float* Wk   = (const float*)d_in[4];
  const float* bk   = (const float*)d_in[5];
  const float* Wv   = (const float*)d_in[6];
  const float* bv   = (const float*)d_in[7];

  int N = in_sizes[0] / IN_FEAT;
  int E = in_sizes[1] / 2;

  float* out   = (float*)d_out;                       // (N, 64)
  float* attn1 = out + (long long)N * EMBED;          // (E, 8)
  float* attn2 = attn1 + (long long)E * N_HEADS;      // (E, 8)

  float* ws        = (float*)d_ws;
  float* Q         = ws;
  float* K         = Q + (long long)N * EMBED;
  float* V         = K + (long long)N * EMBED;
  float* segmax    = V + (long long)N * EMBED;
  float* segsum    = segmax + (long long)N * N_HEADS;
  float* attn_mean = segsum + (long long)N * N_HEADS;

  const int T = 256;
  // init: out = 0, segmax = -inf, segsum = 0
  fill_kernel<<<(N * EMBED + T - 1) / T, T, 0, stream>>>(out, 0.0f, N * EMBED);
  fill_kernel<<<(N * N_HEADS + T - 1) / T, T, 0, stream>>>(segmax, -3.402823466e38f, N * N_HEADS);
  fill_kernel<<<(N * N_HEADS + T - 1) / T, T, 0, stream>>>(segsum, 0.0f, N * N_HEADS);

  int ntiles = (N + 15) / 16;
  qkv_kernel<<<(ntiles + 3) / 4, 128, 0, stream>>>(x, Wq, bq, Wk, bk, Wv, bv,
                                                   Q, K, V, N, ntiles);

  int EH = E * N_HEADS;
  scores_kernel<<<(EH + T - 1) / T, T, 0, stream>>>(edge, Q, K, attn1, segmax, E);
  exp_kernel   <<<(EH + T - 1) / T, T, 0, stream>>>(edge, attn1, segmax, segsum, E);
  norm_kernel  <<<(EH + T - 1) / T, T, 0, stream>>>(edge, attn1, attn2, segsum, attn_mean, E);

  long long EC = (long long)E * EMBED;
  spmm_kernel<<<(int)((EC + T - 1) / T), T, 0, stream>>>(edge, V, attn_mean, out, E);
}
